// SWNet_85237920956828
// MI455X (gfx1250) — matrix operand order repaired and checked by reading, not compiled, not run on hardware
//
#include <hip/hip_runtime.h>
#include <cstdint>

typedef __attribute__((ext_vector_type(16))) _Float16 v16h;
typedef __attribute__((ext_vector_type(8)))  float    v8f;
typedef int v4i_ __attribute__((vector_size(16)));

#ifndef __has_builtin
#define __has_builtin(x) 0
#endif
#if __has_builtin(__builtin_amdgcn_global_load_async_to_lds_b128)
#define ASYNC_LDS 1
#else
#define ASYNC_LDS 0
#endif

#define AS1_ __attribute__((address_space(1)))
#define AS3_ __attribute__((address_space(3)))

// ---------------------------------------------------------------------------
// CDNA5 async tile staging: per-lane 16B global->LDS copy, tracked by ASYNCcnt.
// Fallback: synchronous b128 load + ds_store_b128.
// ---------------------------------------------------------------------------
static __device__ __forceinline__ void async_copy16(_Float16* lds,
                                                    const _Float16* g) {
#if ASYNC_LDS
  __builtin_amdgcn_global_load_async_to_lds_b128((AS1_ v4i_*)g, (AS3_ v4i_*)lds,
                                                 0, 0);
#else
  *(uint4*)lds = *(const uint4*)g;
#endif
}

static __device__ __forceinline__ void zero16(_Float16* lds) {
  uint4 z; z.x = 0u; z.y = 0u; z.z = 0u; z.w = 0u;
  *(uint4*)lds = z;
}

static __device__ __forceinline__ void async_wait_all() {
#if ASYNC_LDS
#if __has_builtin(__builtin_amdgcn_s_wait_asynccnt)
  __builtin_amdgcn_s_wait_asynccnt(0);
#else
  asm volatile("s_wait_asynccnt 0" ::: "memory");
#endif
#endif
}

// ---------------------------------------------------------------------------
// WMMA helpers (gfx1250 wave32).  D(16x16 f32) = A(16x32 f16) * B(32x16 f16) + C
// A: lane<16 row M=lane K=0..15; lane>=16 row M=lane-16 K=16..31
// B: lane<16 col N=lane K=0..15; lane>=16 col N=lane-16 K=16..31
// C/D: VGPR r -> M = r + (lane<16?0:8), N = lane&15
// ---------------------------------------------------------------------------
static __device__ __forceinline__ v8f wmma_16x16x32(v16h a, v16h b, v8f c) {
  return __builtin_amdgcn_wmma_f32_16x16x32_f16(false, a, false, b, (short)0, c,
                                                false, false);
}

static __device__ __forceinline__ v16h load_a_frag(const _Float16* base, int ld,
                                                   int row0, int lane) {
  const _Float16* p = base + (size_t)(row0 + (lane & 15)) * ld + ((lane & 16) ? 16 : 0);
  v16h a;
#pragma unroll
  for (int t = 0; t < 16; ++t) a[t] = p[t];
  return a;
}

// operand stored transposed in LDS: Bt[n][k]
static __device__ __forceinline__ v16h load_bt_frag(const _Float16* base, int ld,
                                                    int col0, int lane) {
  const _Float16* p = base + (size_t)(col0 + (lane & 15)) * ld + ((lane & 16) ? 16 : 0);
  v16h b;
#pragma unroll
  for (int t = 0; t < 16; ++t) b[t] = p[t];
  return b;
}

// operand stored row-major in LDS: B[k][n]
static __device__ __forceinline__ v16h load_b_frag(const _Float16* base, int ld,
                                                   int k0, int col0, int lane) {
  int n  = col0 + (lane & 15);
  int kb = k0 + ((lane & 16) ? 16 : 0);
  v16h b;
#pragma unroll
  for (int t = 0; t < 16; ++t) b[t] = base[(size_t)(kb + t) * ld + n];
  return b;
}

template <int ACT>
static __device__ __forceinline__ float epilogue(float v) {
  if (ACT == 1)  // exact GELU
    return 0.5f * v * (1.0f + erff(v * 0.70710678118654752440f));
  return v;
}

// ---------------------------------------------------------------------------
// f16 GEMM:  Out[M,N] = act(A[M,K] @ W[N,K]^T + bias[N])
// 256 threads = 8 waves (4x2); block tile 128x64; wave tile 32x32 (4 WMMA/K-step)
// A,W are f16 in HBM; tiles staged via async LDS copies. K % 32 == 0.
// ---------------------------------------------------------------------------
template <int ACT, bool OUTF16>
__global__ __launch_bounds__(256) void gemm16_kernel(
    const _Float16* __restrict__ A, const _Float16* __restrict__ Wt,
    const float* __restrict__ bias, void* __restrict__ OutV, int M, int N, int K) {
  __shared__ _Float16 Ah[128][40];
  __shared__ _Float16 Bh[64][40];
  const int tid = threadIdx.x, lane = tid & 31, wid = tid >> 5;
  const int wm = wid >> 1, wn = wid & 1;  // wm: 32-row strip, wn: 32-col strip
  const int m0 = blockIdx.y * 128, n0 = blockIdx.x * 64;
  v8f acc00 = {}, acc01 = {}, acc10 = {}, acc11 = {};
  for (int k0 = 0; k0 < K; k0 += 32) {
    __syncthreads();
#pragma unroll
    for (int i = tid; i < 512; i += 256) {  // A: 128 rows x 4 x 16B
      int r = i >> 2, c = (i & 3) * 8;
      int ga = m0 + r;
      if (ga < M) async_copy16(&Ah[r][c], A + (size_t)ga * K + k0 + c);
      else        zero16(&Ah[r][c]);
    }
    {  // B: 64 rows x 4 x 16B
      int r = tid >> 2, c = (tid & 3) * 8;
      int gb = n0 + r;
      if (gb < N) async_copy16(&Bh[r][c], Wt + (size_t)gb * K + k0 + c);
      else        zero16(&Bh[r][c]);
    }
    async_wait_all();
    __syncthreads();
    v16h a0 = load_a_frag(&Ah[0][0], 40, wm * 32, lane);
    v16h a1 = load_a_frag(&Ah[0][0], 40, wm * 32 + 16, lane);
    v16h b0 = load_bt_frag(&Bh[0][0], 40, wn * 32, lane);
    v16h b1 = load_bt_frag(&Bh[0][0], 40, wn * 32 + 16, lane);
    acc00 = wmma_16x16x32(a0, b0, acc00);
    acc01 = wmma_16x16x32(a0, b1, acc01);
    acc10 = wmma_16x16x32(a1, b0, acc10);
    acc11 = wmma_16x16x32(a1, b1, acc11);
  }
  float* Of = (float*)OutV;
  _Float16* Oh = (_Float16*)OutV;
  const int c0 = n0 + wn * 32 + (lane & 15);
  const int c1 = c0 + 16;
  const int r0 = m0 + wm * 32 + ((lane & 16) ? 8 : 0);
  const float bv0 = (bias && c0 < N) ? bias[c0] : 0.0f;
  const float bv1 = (bias && c1 < N) ? bias[c1] : 0.0f;
#pragma unroll
  for (int half = 0; half < 2; ++half) {
    v8f* p0 = half ? &acc10 : &acc00;
    v8f* p1 = half ? &acc11 : &acc01;
#pragma unroll
    for (int r = 0; r < 8; ++r) {
      int row = r0 + half * 16 + r;
      if (row >= M) continue;
      if (c0 < N) {
        float v = epilogue<ACT>((*p0)[r] + bv0);
        if (OUTF16) Oh[(size_t)row * N + c0] = (_Float16)v;
        else        Of[(size_t)row * N + c0] = v;
      }
      if (c1 < N) {
        float v = epilogue<ACT>((*p1)[r] + bv1);
        if (OUTF16) Oh[(size_t)row * N + c1] = (_Float16)v;
        else        Of[(size_t)row * N + c1] = v;
      }
    }
  }
}

// ---------------------------------------------------------------------------
// Fused window attention (SwinV2 cosine attn), block = (window, head), 128 thr.
// N=49 padded to 64, head_dim=32.  QK^T: 1 WMMA per tile; softmax(+CPB,+mask)
// in LDS f32; P*V: 2 K-steps.  f16 in (qkv), f16 out (for projection GEMM).
// ---------------------------------------------------------------------------
__global__ __launch_bounds__(128) void attn_kernel(
    const _Float16* __restrict__ qkv, const float* __restrict__ logit_scale,
    const float* __restrict__ bias, const float* __restrict__ mask,
    _Float16* __restrict__ out, int C, int nW) {
  const int bw = blockIdx.x, h = blockIdx.y;
  const int tid = threadIdx.x, lane = tid & 31, wid = tid >> 5;
  __shared__ _Float16 Qs[64][40];
  __shared__ _Float16 Ks[64][40];
  __shared__ _Float16 Vs[64][40];
  __shared__ float    Sf[64][65];
  __shared__ _Float16 Ps[64][72];

  for (int i = tid; i < 64 * 40; i += 128) {
    int r = i / 40, c = i % 40;
    Qs[r][c] = (_Float16)0.0f; Ks[r][c] = (_Float16)0.0f; Vs[r][c] = (_Float16)0.0f;
  }
  for (int i = tid; i < 64 * 72; i += 128) Ps[i / 72][i % 72] = (_Float16)0.0f;
  __syncthreads();

  for (int r = tid; r < 49; r += 128) {
    const _Float16* q = qkv + ((size_t)bw * 49 + r) * 3 * C + (size_t)h * 32;
    const _Float16* k = q + C;
    const _Float16* v = q + 2 * C;
    float qs = 0.f, ks = 0.f, qv[32], kv[32];
#pragma unroll
    for (int d = 0; d < 32; ++d) {
      qv[d] = (float)q[d]; kv[d] = (float)k[d];
      qs += qv[d] * qv[d]; ks += kv[d] * kv[d];
    }
    float qi = 1.0f / fmaxf(sqrtf(qs), 1e-12f);
    float ki = 1.0f / fmaxf(sqrtf(ks), 1e-12f);
#pragma unroll
    for (int d = 0; d < 32; ++d) {
      Qs[r][d] = (_Float16)(qv[d] * qi);
      Ks[r][d] = (_Float16)(kv[d] * ki);
      Vs[r][d] = v[d];
    }
  }
  __syncthreads();

  {  // S = Qhat * Khat^T
    v16h a = load_a_frag(&Qs[0][0], 40, wid * 16, lane);
#pragma unroll
    for (int nt = 0; nt < 4; ++nt) {
      v16h b  = load_bt_frag(&Ks[0][0], 40, nt * 16, lane);
      v8f acc = {};
      acc = wmma_16x16x32(a, b, acc);
      int rb = wid * 16 + ((lane & 16) ? 8 : 0);
      int cc = nt * 16 + (lane & 15);
#pragma unroll
      for (int r = 0; r < 8; ++r) Sf[rb + r][cc] = acc[r];
    }
  }
  __syncthreads();

  const float scale = expf(fminf(logit_scale[h], 4.60517018598809f));  // ln(100)
  for (int r = tid; r < 49; r += 128) {
    const float* bp = bias + (size_t)h * 2401 + r * 49;
    const float* mp = mask ? mask + (size_t)(bw % nW) * 2401 + r * 49 : nullptr;
    float mx = -1e30f;
    for (int j = 0; j < 49; ++j) {
      float v = Sf[r][j] * scale + bp[j] + (mp ? mp[j] : 0.0f);
      mx = fmaxf(mx, v);
    }
    float sum = 0.f;
    for (int j = 0; j < 49; ++j) {
      float v = Sf[r][j] * scale + bp[j] + (mp ? mp[j] : 0.0f);
      sum += expf(v - mx);
    }
    float inv = 1.0f / sum;
    for (int j = 0; j < 49; ++j) {
      float v = Sf[r][j] * scale + bp[j] + (mp ? mp[j] : 0.0f);
      Ps[r][j] = (_Float16)(expf(v - mx) * inv);
    }
  }
  __syncthreads();

  {  // O = P * V
#pragma unroll
    for (int ct = 0; ct < 2; ++ct) {
      v8f acc = {};
#pragma unroll
      for (int ks = 0; ks < 2; ++ks) {
        v16h a = load_a_frag(&Ps[0][ks * 32], 72, wid * 16, lane);
        v16h b = load_b_frag(&Vs[0][0], 40, ks * 32, ct * 16, lane);
        acc = wmma_16x16x32(a, b, acc);
      }
      int rb = wid * 16 + ((lane & 16) ? 8 : 0);
      int cc = h * 32 + ct * 16 + (lane & 15);
#pragma unroll
      for (int r = 0; r < 8; ++r) {
        int row = rb + r;
        if (row < 49) out[((size_t)bw * 49 + row) * C + cc] = (_Float16)acc[r];
      }
    }
  }
}

// ---------------------------------------------------------------------------
// LayerNorm.  ADD: dst += LN(src).  Optional f16 shadow of the result.
// ---------------------------------------------------------------------------
template <bool ADD>
__global__ __launch_bounds__(128) void ln_kernel(float* __restrict__ dst,
                                                 const float* __restrict__ src,
                                                 const float* __restrict__ g,
                                                 const float* __restrict__ b,
                                                 _Float16* __restrict__ shadow,
                                                 int C) {
  __shared__ float rs[128], rq[128];
  const int row = blockIdx.x, tid = threadIdx.x;
  const float* s = src + (size_t)row * C;
  float sum = 0.f, sq = 0.f;
  for (int c = tid; c < C; c += 128) { float v = s[c]; sum += v; sq += v * v; }
  rs[tid] = sum; rq[tid] = sq;
  __syncthreads();
  for (int o = 64; o > 0; o >>= 1) {
    if (tid < o) { rs[tid] += rs[tid + o]; rq[tid] += rq[tid + o]; }
    __syncthreads();
  }
  float mean = rs[0] / (float)C;
  float var  = rq[0] / (float)C - mean * mean;
  float inv  = rsqrtf(var + 1e-5f);
  float* d = dst + (size_t)row * C;
  for (int c = tid; c < C; c += 128) {
    float v = (s[c] - mean) * inv * g[c] + b[c];
    float o = ADD ? (d[c] + v) : v;
    d[c] = o;
    if (shadow) shadow[(size_t)row * C + c] = (_Float16)o;
  }
}

// ---------------------------------------------------------------------------
// Data-movement / setup kernels
// ---------------------------------------------------------------------------
__global__ void cvt_f32_f16_kernel(const float* __restrict__ s,
                                   _Float16* __restrict__ d, size_t n) {
  size_t i = (size_t)blockIdx.x * blockDim.x + threadIdx.x;
  if (i < n) d[i] = (_Float16)s[i];
}

__global__ void patch_embed_kernel(const float* __restrict__ x,
                                   const float* __restrict__ pw,
                                   const float* __restrict__ pb,
                                   float* __restrict__ out, size_t total) {
  size_t i = (size_t)blockIdx.x * blockDim.x + threadIdx.x;
  if (i >= total) return;
  int e = (int)(i % 96);
  size_t t = i / 96;
  int l = (int)(t % 3136);
  int b = (int)(t / 3136);
  int py = l / 56, px = l % 56;
  float acc = pb[e];
  const float* w = pw + (size_t)e * 48;
  for (int c = 0; c < 3; ++c)
    for (int ii = 0; ii < 4; ++ii)
      for (int jj = 0; jj < 4; ++jj)
        acc += x[(((size_t)b * 3 + c) * 224 + py * 4 + ii) * 224 + px * 4 + jj] *
               w[c * 16 + ii * 4 + jj];
  out[i] = acc;
}

__global__ void window_partition_kernel(const float* __restrict__ t,
                                        _Float16* __restrict__ o, int H, int W,
                                        int C, int shift, size_t total) {
  size_t i = (size_t)blockIdx.x * blockDim.x + threadIdx.x;
  if (i >= total) return;
  int c = (int)(i % C);
  size_t r = i / C;
  int n = (int)(r % 49);
  size_t bw = r / 49;
  int nWw = W / 7, nW = (H / 7) * nWw;
  int b = (int)(bw / nW), win = (int)(bw % nW);
  int wh = win / nWw, ww = win % nWw;
  int hh = (wh * 7 + n / 7 + shift) % H;
  int wp = (ww * 7 + n % 7 + shift) % W;
  o[i] = (_Float16)t[((size_t)b * (H * W) + hh * W + wp) * C + c];
}

__global__ void window_reverse_kernel(const float* __restrict__ wbuf,
                                      float* __restrict__ o, int H, int W, int C,
                                      int shift, size_t total) {
  size_t i = (size_t)blockIdx.x * blockDim.x + threadIdx.x;
  if (i >= total) return;
  int c = (int)(i % C);
  size_t r = i / C;
  int l = (int)(r % (H * W));
  int b = (int)(r / (H * W));
  int h = l / W, w = l % W;
  int hp = (h - shift + H) % H, wp = (w - shift + W) % W;
  int nWw = W / 7, nW = (H / 7) * nWw;
  int wh = hp / 7, ii = hp % 7, ww = wp / 7, jj = wp % 7;
  size_t bw = (size_t)b * nW + wh * nWw + ww;
  o[i] = wbuf[(bw * 49 + ii * 7 + jj) * C + c];
}

__global__ void make_qkv_bias_kernel(const float* __restrict__ qb,
                                     const float* __restrict__ vb,
                                     float* __restrict__ o, int C) {
  int i = blockIdx.x * blockDim.x + threadIdx.x;
  if (i >= 3 * C) return;
  o[i] = (i < C) ? qb[i] : (i < 2 * C ? 0.0f : vb[i - 2 * C]);
}

__global__ __launch_bounds__(192) void cpb_table_kernel(
    const float* __restrict__ w1, const float* __restrict__ b1,
    const float* __restrict__ w2, float* __restrict__ tbl) {
  int h = blockIdx.x, pos = threadIdx.x;
  if (pos >= 169) return;
  int di = pos / 13 - 6, dj = pos % 13 - 6;
  auto enc = [](int d) -> float {
    float t = (float)d * (8.0f / 6.0f);
    float a = log2f(fabsf(t) + 1.0f) * (1.0f / 3.0f);  // / log2(8)
    return t < 0.0f ? -a : a;
  };
  float t0 = enc(di), t1 = enc(dj);
  const float* w2h = w2 + (size_t)h * 512;
  float acc = 0.f;
  for (int k = 0; k < 512; ++k) {
    float hv = fmaxf(t0 * w1[2 * k] + t1 * w1[2 * k + 1] + b1[k], 0.0f);
    acc += hv * w2h[k];
  }
  tbl[(size_t)h * 169 + pos] = 16.0f / (1.0f + expf(-acc));
}

__global__ void bias_expand_kernel(const float* __restrict__ tbl,
                                   float* __restrict__ bias, int nh) {
  int i = blockIdx.x * blockDim.x + threadIdx.x;
  if (i >= nh * 2401) return;
  int h = i / 2401, rem = i % 2401;
  int qi = rem / 49, kj = rem % 49;
  int idx = (qi / 7 - kj / 7 + 6) * 13 + (qi % 7 - kj % 7 + 6);
  bias[i] = tbl[h * 169 + idx];
}

__global__ void mask_kernel(float* __restrict__ mask, int H, int W, int s,
                            int total) {
  int i = blockIdx.x * blockDim.x + threadIdx.x;
  if (i >= total) return;
  int win = i / 2401, rem = i % 2401;
  int qi = rem / 49, kj = rem % 49;
  int nWw = W / 7;
  int wh = win / nWw, ww = win % nWw;
  auto reg = [](int v, int D, int sh) -> int {
    return v < D - 7 ? 0 : (v < D - sh ? 1 : 2);
  };
  int idq = 3 * reg(wh * 7 + qi / 7, H, s) + reg(ww * 7 + qi % 7, W, s);
  int idk = 3 * reg(wh * 7 + kj / 7, H, s) + reg(ww * 7 + kj % 7, W, s);
  mask[i] = (idq != idk) ? -100.0f : 0.0f;
}

__global__ void merge_gather_kernel(const float* __restrict__ t,
                                    _Float16* __restrict__ o, int H, int W, int C,
                                    size_t total) {
  size_t i = (size_t)blockIdx.x * blockDim.x + threadIdx.x;
  if (i >= total) return;
  int C4 = 4 * C;
  int c4 = (int)(i % C4);
  size_t r = i / C4;
  int L2 = (H / 2) * (W / 2);
  int l2 = (int)(r % L2);
  int b = (int)(r / L2);
  int qd = c4 / C, c = c4 % C;
  int h2 = l2 / (W / 2), w2 = l2 % (W / 2);
  int dh = (qd == 1 || qd == 3) ? 1 : 0;
  int dw = (qd >= 2) ? 1 : 0;
  o[i] = (_Float16)t[((size_t)b * (H * W) + (2 * h2 + dh) * W + (2 * w2 + dw)) * C + c];
}

__global__ void copy_kernel(const float* __restrict__ s, float* __restrict__ d,
                            size_t n) {
  size_t i = (size_t)blockIdx.x * blockDim.x + threadIdx.x;
  if (i < n) d[i] = s[i];
}

// ---------------------------------------------------------------------------
// Host orchestration
// ---------------------------------------------------------------------------
struct BlkP {
  const float *qkv_w, *q_bias, *v_bias, *logit_scale, *cpb_w1, *cpb_b1, *cpb_w2,
      *proj_w, *proj_b, *n1g, *n1b, *n2g, *n2b, *fc1_w, *fc1_b, *fc2_w, *fc2_b;
};
struct MrgP { const float *red_w, *ng, *nb; };

static inline int cdiv(long long a, long long b) { return (int)((a + b - 1) / b); }

extern "C" void kernel_launch(void* const* d_in, const int* in_sizes, int n_in,
                              void* d_out, int out_size, void* d_ws, size_t ws_size,
                              hipStream_t stream) {
  (void)in_sizes; (void)n_in; (void)ws_size;
  static const int HEADS_[4] = {3, 6, 12, 24};
  int ix = 0;
  auto F = [&](int i) { return (const float*)d_in[i]; };
  const float* x        = F(ix++);
  const float* patch_w  = F(ix++);
  const float* patch_b  = F(ix++);
  const float* patch_ng = F(ix++);
  const float* patch_nb = F(ix++);
  BlkP blk[4][2];
  MrgP mg[3];
  for (int s = 0; s < 4; ++s) {
    for (int j = 0; j < 2; ++j) {
      BlkP& b = blk[s][j];
      b.qkv_w = F(ix++); b.q_bias = F(ix++); b.v_bias = F(ix++);
      b.logit_scale = F(ix++);
      b.cpb_w1 = F(ix++); b.cpb_b1 = F(ix++); b.cpb_w2 = F(ix++);
      b.proj_w = F(ix++); b.proj_b = F(ix++);
      b.n1g = F(ix++); b.n1b = F(ix++); b.n2g = F(ix++); b.n2b = F(ix++);
      b.fc1_w = F(ix++); b.fc1_b = F(ix++); b.fc2_w = F(ix++); b.fc2_b = F(ix++);
    }
    if (s < 3) { mg[s].red_w = F(ix++); mg[s].ng = F(ix++); mg[s].nb = F(ix++); }
  }

  // Workspace layout.  SZ = B*L1*C1 = 32*3136*96 elements.
  const size_t SZ = (size_t)32 * 3136 * 96;
  float*    T   = (float*)d_ws;        // persistent tokens (f32)
  float*    F1  = T + SZ;              // f32 scratch
  float*    F2  = F1 + SZ;             // f32 scratch
  _Float16* XWh = (_Float16*)(F2 + SZ);  // windowed input / merge gather (f16)
  _Float16* AOh = XWh + SZ;              // attention output (f16)
  _Float16* Th  = AOh + SZ;              // f16 shadow of tokens (MLP input)
  _Float16* R1h = Th + SZ;               // qkv (3*SZ) / MLP hidden (4*SZ), f16
  _Float16* Wh  = R1h + 4 * SZ;          // f16 weight arena (~21M halves)
  float* SMALL  = (float*)(Wh + (size_t)24 * 1024 * 1024);
  float* QKVB = SMALL;
  float* TBL  = SMALL + 4096;
  float* BIAS = SMALL + 16384;
  float* MASK = SMALL + 90112;

  // One-time (per launch) weight conversion to f16.
  _Float16* wp = Wh;
  auto CVT = [&](const float* w, size_t n) {
    _Float16* p = wp;
    wp += (n + 7) & ~(size_t)7;
    cvt_f32_f16_kernel<<<cdiv(n, 256), 256, 0, stream>>>(w, p, n);
    return (const _Float16*)p;
  };
  const _Float16 *qkvw16[4][2], *projw16[4][2], *fc1w16[4][2], *fc2w16[4][2],
      *redw16[3];
  {
    int C = 96;
    for (int s = 0; s < 4; ++s) {
      for (int j = 0; j < 2; ++j) {
        qkvw16[s][j]  = CVT(blk[s][j].qkv_w, (size_t)3 * C * C);
        projw16[s][j] = CVT(blk[s][j].proj_w, (size_t)C * C);
        fc1w16[s][j]  = CVT(blk[s][j].fc1_w, (size_t)4 * C * C);
        fc2w16[s][j]  = CVT(blk[s][j].fc2_w, (size_t)4 * C * C);
      }
      if (s < 3) redw16[s] = CVT(mg[s].red_w, (size_t)8 * C * C);
      C *= 2;
    }
  }

  // Patch embed + LN
  patch_embed_kernel<<<cdiv(SZ, 256), 256, 0, stream>>>(x, patch_w, patch_b, F1, SZ);
  ln_kernel<false><<<32 * 3136, 128, 0, stream>>>(T, F1, patch_ng, patch_nb,
                                                  nullptr, 96);

  int Hc = 56, Wc = 56, C = 96;
  for (int s = 0; s < 4; ++s) {
    const int L = Hc * Wc, nh = HEADS_[s];
    for (int j = 0; j < 2; ++j) {
      const BlkP& bp = blk[s][j];
      int shift = (j & 1) ? 3 : 0;
      if (Hc <= 7) shift = 0;  // last stage: window == feature map
      const int nWw = Wc / 7, nW = (Hc / 7) * nWw;
      const int Bw = 32 * nW, M = Bw * 49;
      const size_t we = (size_t)M * C;

      cpb_table_kernel<<<nh, 192, 0, stream>>>(bp.cpb_w1, bp.cpb_b1, bp.cpb_w2, TBL);
      bias_expand_kernel<<<cdiv(nh * 2401, 256), 256, 0, stream>>>(TBL, BIAS, nh);
      if (shift)
        mask_kernel<<<cdiv(nW * 2401, 256), 256, 0, stream>>>(MASK, Hc, Wc, shift,
                                                              nW * 2401);
      make_qkv_bias_kernel<<<cdiv(3 * C, 256), 256, 0, stream>>>(bp.q_bias,
                                                                 bp.v_bias, QKVB, C);
      window_partition_kernel<<<cdiv(we, 256), 256, 0, stream>>>(T, XWh, Hc, Wc, C,
                                                                 shift, we);
      {  // qkv projection -> f16
        dim3 g(cdiv(3 * C, 64), cdiv(M, 128));
        gemm16_kernel<0, true><<<g, 256, 0, stream>>>(XWh, qkvw16[s][j], QKVB, R1h,
                                                      M, 3 * C, C);
      }
      attn_kernel<<<dim3(Bw, nh), 128, 0, stream>>>(R1h, bp.logit_scale, BIAS,
                                                    shift ? MASK : nullptr, AOh, C,
                                                    nW > 0 ? nW : 1);
      {  // output projection -> f32
        dim3 g(cdiv(C, 64), cdiv(M, 128));
        gemm16_kernel<0, false><<<g, 256, 0, stream>>>(AOh, projw16[s][j],
                                                       bp.proj_b, F1, M, C, C);
      }
      window_reverse_kernel<<<cdiv(we, 256), 256, 0, stream>>>(F1, F2, Hc, Wc, C,
                                                               shift, we);
      ln_kernel<true><<<32 * L, 128, 0, stream>>>(T, F2, bp.n1g, bp.n1b, Th, C);
      {  // MLP
        dim3 g1(cdiv(4 * C, 64), cdiv(32 * L, 128));
        gemm16_kernel<1, true><<<g1, 256, 0, stream>>>(Th, fc1w16[s][j], bp.fc1_b,
                                                       R1h, 32 * L, 4 * C, C);
        dim3 g2(cdiv(C, 64), cdiv(32 * L, 128));
        gemm16_kernel<0, false><<<g2, 256, 0, stream>>>(R1h, fc2w16[s][j],
                                                        bp.fc2_b, F1, 32 * L, C,
                                                        4 * C);
      }
      ln_kernel<true><<<32 * L, 128, 0, stream>>>(T, F1, bp.n2g, bp.n2b, nullptr, C);
    }
    if (s < 3) {  // patch merging
      const int L2 = L / 4, M3 = 32 * L2, C4 = 4 * C, C2 = 2 * C;
      size_t tot = (size_t)M3 * C4;
      merge_gather_kernel<<<cdiv(tot, 256), 256, 0, stream>>>(T, XWh, Hc, Wc, C, tot);
      dim3 g(cdiv(C2, 64), cdiv(M3, 128));
      gemm16_kernel<0, false><<<g, 256, 0, stream>>>(XWh, redw16[s], nullptr, F1,
                                                     M3, C2, C4);
      ln_kernel<false><<<M3, 128, 0, stream>>>(T, F1, mg[s].ng, mg[s].nb, nullptr,
                                               C2);
      Hc /= 2; Wc /= 2; C *= 2;
    }
  }

  copy_kernel<<<cdiv((size_t)out_size, 256), 256, 0, stream>>>(T, (float*)d_out,
                                                               (size_t)out_size);
}
